// CausalMultiheadRoPEAttn_69329362092087
// MI455X (gfx1250) — matrix operand order repaired and checked by reading, not compiled, not run on hardware
//
#include <hip/hip_runtime.h>
#include <hip/hip_bf16.h>
#include <math.h>

#define D_MODEL 1024
#define NUM_HEADS 16
#define D_HEAD 64
#define SEQ 2048
#define BATCH 2
#define M_TOK (BATCH * SEQ)
#define SCALE 0.125f /* 1/sqrt(64) */

typedef __attribute__((ext_vector_type(16))) __bf16 bf16x16;
typedef __attribute__((ext_vector_type(8)))  float  v8f;
typedef __attribute__((ext_vector_type(4)))  unsigned u32x4;

union FragU { bf16x16 v; u32x4 q[2]; };

__device__ __forceinline__ unsigned short f2bf(float f) {
    unsigned u = __builtin_bit_cast(unsigned, f);
    u += 0x7FFFu + ((u >> 16) & 1u);          // round-to-nearest-even
    return (unsigned short)(u >> 16);
}
__device__ __forceinline__ unsigned pk2(float a, float b) {
    return (unsigned)f2bf(a) | ((unsigned)f2bf(b) << 16);
}
__device__ __forceinline__ v8f vzero8() {
    v8f z; for (int i = 0; i < 8; i++) z[i] = 0.0f; return z;
}
__device__ __forceinline__ v8f wmma_bf16(bf16x16 a, bf16x16 b, v8f c) {
    return __builtin_amdgcn_wmma_f32_16x16x32_bf16(
        /*neg_a=*/false, a, /*neg_b=*/false, b,
        /*c_mod=*/(short)0, c, /*reuse_a=*/false, /*reuse_b=*/false);
}

// ---------------------------------------------------------------------------
// GEMM: Y[M,N] = X[M,K] @ W[N,K]^T + bias[N]   (fp32 in/out, bf16 WMMA core)
// Block: 256 threads = 8 waves, tile 128(M) x 128(N), BK = 32.
// Wave (wr=wid&3, wc=wid>>2) owns a 32x64 strip: 2 A-frags x 4 B-frags
// -> 8 WMMAs per K-step, all fragments loaded before the WMMA chain so the
// scheduler can overlap ds_load with v_wmma instead of s_wait_dscnt 0.
// ---------------------------------------------------------------------------
#define BM 128
#define BN 128
#define BK 32

__global__ __launch_bounds__(256)
void gemm_xwT_bias(const float* __restrict__ X, const float* __restrict__ W,
                   const float* __restrict__ bias, float* __restrict__ Y,
                   int M, int N, int K)
{
    __shared__ unsigned short As[BM][BK + 8];   // row stride 40 halves = 80B
    __shared__ unsigned short Bs[BN][BK + 8];

    const int tid  = threadIdx.x;
    const int lane = tid & 31;
    const int wid  = tid >> 5;
    const int wr   = wid & 3;        // M group (32 rows)
    const int wc   = wid >> 2;       // N group (64 cols)
    const int hi   = lane >> 4;
    const int l16  = lane & 15;
    const int m0   = blockIdx.y * BM;
    const int n0   = blockIdx.x * BN;

    v8f acc[2][4];
    for (int m = 0; m < 2; m++)
        for (int t = 0; t < 4; t++) acc[m][t] = vzero8();

    for (int k0 = 0; k0 < K; k0 += BK) {
        __syncthreads();
        // stage A: 128x32 f32 -> bf16 (1024 float4, 4 per thread)
        for (int i = 0; i < 4; i++) {
            int lin = tid + i * 256;
            int r   = lin >> 3;
            int c4  = (lin & 7) << 2;
            const float* src = &X[(size_t)(m0 + r) * K + k0 + c4];
            float4 v = *(const float4*)src;
            unsigned* dst = (unsigned*)&As[r][c4];
            dst[0] = pk2(v.x, v.y);
            dst[1] = pk2(v.z, v.w);
            if (k0 + BK < K) __builtin_prefetch(src + BK, 0, 0);
        }
        // stage B: 128x32 f32 -> bf16 (1024 float4, 4 per thread)
        for (int i = 0; i < 4; i++) {
            int lin = tid + i * 256;
            int r   = lin >> 3;
            int c4  = (lin & 7) << 2;
            const float* src = &W[(size_t)(n0 + r) * K + k0 + c4];
            float4 v = *(const float4*)src;
            unsigned* dst = (unsigned*)&Bs[r][c4];
            dst[0] = pk2(v.x, v.y);
            dst[1] = pk2(v.z, v.w);
            if (k0 + BK < K) __builtin_prefetch(src + BK, 0, 0);
        }
        __syncthreads();

        // load ALL fragments first, then issue the 8-WMMA chain
        FragU ua[2];
        for (int m = 0; m < 2; m++) {
            const unsigned short* arow = &As[wr * 32 + m * 16 + l16][0];
            ua[m].q[0] = *(const u32x4*)(arow + hi * 8);
            ua[m].q[1] = *(const u32x4*)(arow + hi * 8 + 16);
        }
        FragU ub[4];
        for (int t = 0; t < 4; t++) {
            const unsigned short* brow = &Bs[wc * 64 + t * 16 + l16][0];
            ub[t].q[0] = *(const u32x4*)(brow + hi * 8);
            ub[t].q[1] = *(const u32x4*)(brow + hi * 8 + 16);
        }
        for (int m = 0; m < 2; m++)
            for (int t = 0; t < 4; t++)
                acc[m][t] = wmma_bf16(ua[m].v, ub[t].v, acc[m][t]);
    }

    // epilogue: C layout M = r + 8*hi, N = l16
    for (int m = 0; m < 2; m++) {
        for (int t = 0; t < 4; t++) {
            int col = n0 + wc * 64 + t * 16 + l16;
            float bv = bias[col];
            for (int r = 0; r < 8; r++) {
                int row = m0 + wr * 32 + m * 16 + r + 8 * hi;
                Y[(size_t)row * N + col] = acc[m][t][r] + bv;
            }
        }
    }
}

// ---------------------------------------------------------------------------
// RoPE applied in-place to Q and K ([M_TOK, D_MODEL] fp32). One thread per
// (token, pair) = 4096 * 512 threads.
// ---------------------------------------------------------------------------
__global__ __launch_bounds__(256)
void rope_kernel(float* __restrict__ Q, float* __restrict__ K)
{
    int p = blockIdx.x * blockDim.x + threadIdx.x;   // 0 .. M_TOK*512-1
    int row  = p >> 9;
    int cp   = p & 511;
    int head = cp >> 5;
    int i    = cp & 31;
    int col0 = head * D_HEAD + 2 * i;
    int pos  = row & (SEQ - 1);

    // inv_freq = theta^{-2i/64} = exp(-i * ln(10000)/32)
    float freq = __expf(-0.28782314f * (float)i);
    float ang  = (float)pos * freq;
    float s, c;
    __sincosf(ang, &s, &c);

    size_t idx = (size_t)row * D_MODEL + col0;
    float q0 = Q[idx], q1 = Q[idx + 1];
    Q[idx]     = q0 * c - q1 * s;
    Q[idx + 1] = q0 * s + q1 * c;
    float k0 = K[idx], k1 = K[idx + 1];
    K[idx]     = k0 * c - k1 * s;
    K[idx + 1] = k0 * s + k1 * c;
}

// ---------------------------------------------------------------------------
// Causal flash attention. Grid (SEQ/128, NUM_HEADS, BATCH), 256 threads.
// Each wave owns 16 query rows; K/V streamed in 32-key tiles via LDS (bf16).
// ---------------------------------------------------------------------------
__global__ __launch_bounds__(256)
void attn_kernel(const float* __restrict__ Q, const float* __restrict__ Kb,
                 const float* __restrict__ Vb, float* __restrict__ O)
{
    constexpr int QT = 128;
    __shared__ unsigned short Qs[QT][D_HEAD + 8];      // [q][d]  stride 72
    __shared__ unsigned short Ks[32][D_HEAD + 8];      // [k][d]  stride 72
    __shared__ unsigned short Vs[D_HEAD][32 + 8];      // [d][k]  stride 40 (transposed)
    __shared__ unsigned short Ps[8][16][40];           // per-wave P scratch

    const int tid  = threadIdx.x;
    const int lane = tid & 31;
    const int wid  = tid >> 5;
    const int hi   = lane >> 4;
    const int l16  = lane & 15;

    const int qbase   = blockIdx.x * QT;
    const int cbase   = blockIdx.y * D_HEAD;
    const int rowbase = blockIdx.z * SEQ;

    // stage Q tile 128x64 f32 -> bf16 (2048 float4, 8 per thread)
    for (int i = 0; i < 8; i++) {
        int lin = tid + i * 256;
        int r   = lin >> 4;
        int c4  = (lin & 15) << 2;
        float4 v = *(const float4*)&Q[(size_t)(rowbase + qbase + r) * D_MODEL + cbase + c4];
        unsigned* dst = (unsigned*)&Qs[r][c4];
        dst[0] = pk2(v.x, v.y);
        dst[1] = pk2(v.z, v.w);
    }
    __syncthreads();

    // resident Q A-fragments, dblk = 0 (d 0..31), 1 (d 32..63)
    bf16x16 qa[2];
    {
        const unsigned short* qrow = &Qs[wid * 16 + l16][0];
        for (int d = 0; d < 2; d++) {
            FragU u;
            u.q[0] = *(const u32x4*)(qrow + d * 32 + hi * 8);
            u.q[1] = *(const u32x4*)(qrow + d * 32 + hi * 8 + 16);
            qa[d] = u.v;
        }
    }

    float mrow[8], lrow[8];
    v8f oacc[4];
    for (int r = 0; r < 8; r++) { mrow[r] = -3.0e38f; lrow[r] = 0.0f; }
    for (int t = 0; t < 4; t++) oacc[t] = vzero8();

    const int qrow_max = qbase + wid * 16 + 15;
    const int kend     = qbase + QT;     // causal upper bound, <= SEQ

    for (int kbase = 0; kbase < kend; kbase += 32) {
        __syncthreads();
        // stage K tile 32x64 (512 float4, 2/thread)
        for (int i = 0; i < 2; i++) {
            int lin = tid + i * 256;
            int r   = lin >> 4;
            int c4  = (lin & 15) << 2;
            float4 v = *(const float4*)&Kb[(size_t)(rowbase + kbase + r) * D_MODEL + cbase + c4];
            unsigned* dst = (unsigned*)&Ks[r][c4];
            dst[0] = pk2(v.x, v.y);
            dst[1] = pk2(v.z, v.w);
        }
        // stage V tile transposed -> Vs[d][k]
        for (int i = 0; i < 2; i++) {
            int lin = tid + i * 256;
            int r   = lin >> 4;
            int c4  = (lin & 15) << 2;
            float4 v = *(const float4*)&Vb[(size_t)(rowbase + kbase + r) * D_MODEL + cbase + c4];
            Vs[c4 + 0][r] = f2bf(v.x);
            Vs[c4 + 1][r] = f2bf(v.y);
            Vs[c4 + 2][r] = f2bf(v.z);
            Vs[c4 + 3][r] = f2bf(v.w);
        }
        __syncthreads();

        if (kbase > qrow_max) continue;   // wave-uniform: fully masked tile

        // scores: load all 4 K-fragments, then 4 WMMAs
        FragU kb[2][2];
        for (int c = 0; c < 2; c++) {
            const unsigned short* krow = &Ks[c * 16 + l16][0];
            for (int d = 0; d < 2; d++) {
                kb[c][d].q[0] = *(const u32x4*)(krow + d * 32 + hi * 8);
                kb[c][d].q[1] = *(const u32x4*)(krow + d * 32 + hi * 8 + 16);
            }
        }
        v8f sc[2];
        for (int c = 0; c < 2; c++) {
            v8f s = vzero8();
            s = wmma_bf16(qa[0], kb[c][0].v, s);
            s = wmma_bf16(qa[1], kb[c][1].v, s);
            sc[c] = s;
        }

        // mask + online softmax (16-lane reductions stay within each half)
        for (int r = 0; r < 8; r++) {
            int row = qbase + wid * 16 + r + 8 * hi;
            float s0 = (kbase + l16      <= row) ? sc[0][r] * SCALE : -1.0e30f;
            float s1 = (kbase + 16 + l16 <= row) ? sc[1][r] * SCALE : -1.0e30f;
            float tmax = fmaxf(s0, s1);
            for (int off = 1; off < 16; off <<= 1)
                tmax = fmaxf(tmax, __shfl_xor(tmax, off, 32));
            float mnew  = fmaxf(mrow[r], tmax);
            float alpha = __expf(mrow[r] - mnew);
            float p0 = __expf(s0 - mnew);
            float p1 = __expf(s1 - mnew);
            float rs = p0 + p1;
            for (int off = 1; off < 16; off <<= 1)
                rs += __shfl_xor(rs, off, 32);
            lrow[r] = lrow[r] * alpha + rs;
            mrow[r] = mnew;
            for (int t = 0; t < 4; t++) oacc[t][r] *= alpha;
            Ps[wid][r + 8 * hi][l16]      = f2bf(p0);
            Ps[wid][r + 8 * hi][16 + l16] = f2bf(p1);
        }

        __builtin_amdgcn_wave_barrier();   // keep same-wave LDS st->ld in order

        // P as A-fragment + all V fragments, then the 4-WMMA O update
        FragU up;
        const unsigned short* prow = &Ps[wid][l16][0];
        up.q[0] = *(const u32x4*)(prow + hi * 8);
        up.q[1] = *(const u32x4*)(prow + hi * 8 + 16);
        FragU uv[4];
        for (int t = 0; t < 4; t++) {
            const unsigned short* vrow = &Vs[t * 16 + l16][0];
            uv[t].q[0] = *(const u32x4*)(vrow + hi * 8);
            uv[t].q[1] = *(const u32x4*)(vrow + hi * 8 + 16);
        }
        for (int t = 0; t < 4; t++)
            oacc[t] = wmma_bf16(up.v, uv[t].v, oacc[t]);
    }

    // finalize: O / l
    for (int t = 0; t < 4; t++) {
        for (int r = 0; r < 8; r++) {
            int row = qbase + wid * 16 + r + 8 * hi;
            O[(size_t)(rowbase + row) * D_MODEL + cbase + t * 16 + l16] =
                oacc[t][r] / lrow[r];
        }
    }
}

// ---------------------------------------------------------------------------
extern "C" void kernel_launch(void* const* d_in, const int* in_sizes, int n_in,
                              void* d_out, int out_size, void* d_ws, size_t ws_size,
                              hipStream_t stream)
{
    const float* qkv = (const float*)d_in[0];
    const float* Wq  = (const float*)d_in[1];
    const float* bq  = (const float*)d_in[2];
    const float* Wk  = (const float*)d_in[3];
    const float* bk  = (const float*)d_in[4];
    const float* Wv  = (const float*)d_in[5];
    const float* bv  = (const float*)d_in[6];
    const float* Wo  = (const float*)d_in[7];
    const float* bo  = (const float*)d_in[8];

    const size_t bufElems = (size_t)M_TOK * D_MODEL;   // 4 Mi floats = 16 MB
    float* Qbuf = (float*)d_ws;
    float* Kbuf = Qbuf + bufElems;
    float* Vbuf = Kbuf + bufElems;
    float* Abuf = Vbuf + bufElems;

    dim3 gGemm(D_MODEL / BN, M_TOK / BM);              // (8, 32)

    gemm_xwT_bias<<<gGemm, 256, 0, stream>>>(qkv, Wq, bq, Qbuf, M_TOK, D_MODEL, D_MODEL);
    gemm_xwT_bias<<<gGemm, 256, 0, stream>>>(qkv, Wk, bk, Kbuf, M_TOK, D_MODEL, D_MODEL);
    gemm_xwT_bias<<<gGemm, 256, 0, stream>>>(qkv, Wv, bv, Vbuf, M_TOK, D_MODEL, D_MODEL);

    int ropeThreads = M_TOK * (D_MODEL / 2);           // 2M
    rope_kernel<<<ropeThreads / 256, 256, 0, stream>>>(Qbuf, Kbuf);

    attn_kernel<<<dim3(SEQ / 128, NUM_HEADS, BATCH), 256, 0, stream>>>(
        Qbuf, Kbuf, Vbuf, Abuf);

    gemm_xwT_bias<<<gGemm, 256, 0, stream>>>(Abuf, Wo, bo, (float*)d_out,
                                             M_TOK, D_MODEL, D_MODEL);
}
